// S2ConvNet_Autoencoder_27376121545244
// MI455X (gfx1250) — matrix operand order repaired and checked by reading, not compiled, not run on hardware
//
#include <hip/hip_runtime.h>

// ---------------------------------------------------------------------------
// Types
// ---------------------------------------------------------------------------
typedef __attribute__((ext_vector_type(4)))  unsigned int v4u;
typedef __attribute__((ext_vector_type(8)))  unsigned int v8u;
typedef __attribute__((ext_vector_type(2)))  float        v2f;
typedef __attribute__((ext_vector_type(8)))  float        v8f;
typedef __attribute__((ext_vector_type(16))) __bf16       v16bf;

// ---------------------------------------------------------------------------
// Problem constants
// ---------------------------------------------------------------------------
#define NAr   22
#define NBr   11
#define NCr   22
#define QTOT  5324          // NA*NB*NC
#define QPAD  5376          // 84 * 64
#define DTOT  165           // sum (2l+1)^2, l<=4
#define BPADC 176           // DTOT padded to 11*16
#define TROWS 192           // DTOT padded to 6*32
#define QT    64
#define NQT   (QPAD/QT)     // 84
#define BATCHC 32
#define PI_D 3.14159265358979323846

// workspace layout (bytes, all 16B aligned)
constexpr size_t OFF_WQ  = 0;
constexpr size_t OFF_BHI = 21504;                                   // QPAD*4
constexpr size_t OFF_BLO = OFF_BHI + (size_t)QPAD * BPADC * 2;
constexpr size_t OFF_THI = OFF_BLO + (size_t)QPAD * BPADC * 2;
constexpr size_t OFF_TLO = OFF_THI + (size_t)TROWS * QPAD * 2;
constexpr size_t OFF_XHI = OFF_TLO + (size_t)TROWS * QPAD * 2;
constexpr size_t OFF_XLO = OFF_XHI + (size_t)4096 * 192 * 2;
constexpr size_t OFF_A   = OFF_XLO + (size_t)4096 * 192 * 2;
constexpr size_t OFF_B   = OFF_A   + (size_t)4096 * DTOT * 4;

// ---------------------------------------------------------------------------
// bf16 split helpers
// ---------------------------------------------------------------------------
__device__ inline unsigned short bf16_rn(float f) {
    unsigned int u = __float_as_uint(f);
    unsigned int r = u + 0x7FFFu + ((u >> 16) & 1u);
    return (unsigned short)(r >> 16);
}
__device__ inline float bf16f(unsigned short h) {
    return __uint_as_float(((unsigned int)h) << 16);
}
__device__ inline void splitpack(float x0, float x1, unsigned int& hw, unsigned int& lw) {
    unsigned short h0 = bf16_rn(x0), h1 = bf16_rn(x1);
    unsigned short l0 = bf16_rn(x0 - bf16f(h0));
    unsigned short l1 = bf16_rn(x1 - bf16f(h1));
    hw = (unsigned int)h0 | ((unsigned int)h1 << 16);
    lw = (unsigned int)l0 | ((unsigned int)l1 << 16);
}
__device__ inline v16bf mk_frag(v4u a, v4u b) {
    v8u t;
    t[0] = a[0]; t[1] = a[1]; t[2] = a[2]; t[3] = a[3];
    t[4] = b[0]; t[5] = b[1]; t[6] = b[2]; t[7] = b[3];
    return __builtin_bit_cast(v16bf, t);
}
__device__ inline v8f zero8() { v8f z; for (int i = 0; i < 8; ++i) z[i] = 0.f; return z; }
__device__ inline v8f wmma_bf16(v16bf a, v16bf b, v8f c) {
    return __builtin_amdgcn_wmma_f32_16x16x32_bf16(false, a, false, b, (short)0, c, false, false);
}

// ---------------------------------------------------------------------------
// Basis builder: reproduces _get_basis on device (Wigner D real basis)
// ---------------------------------------------------------------------------
__device__ inline double ipow_d(double b, int e) {
    double r = 1.0;
    for (int i = 0; i < e; ++i) r *= b;
    return r;
}
// sparse real-U row: returns #nonzeros, fills mu (actual m value), complex coef
__device__ inline int urow(int mr, int* mu, double* cr, double* ci) {
    const double INVR2 = 0.70710678118654752440;
    if (mr == 0) { mu[0] = 0; cr[0] = 1.0; ci[0] = 0.0; return 1; }
    if (mr > 0) {
        mu[0] = mr;  cr[0] = (mr & 1) ? -INVR2 : INVR2; ci[0] = 0.0;
        mu[1] = -mr; cr[1] = INVR2;                      ci[1] = 0.0;
        return 2;
    }
    int a = -mr;
    mu[0] = mr; cr[0] = 0.0; ci[0] = INVR2;
    mu[1] = a;  cr[1] = 0.0; ci[1] = -((a & 1) ? -INVR2 : INVR2);
    return 2;
}

__global__ void build_basis_kernel(unsigned short* __restrict__ Bhi,
                                   unsigned short* __restrict__ Blo,
                                   unsigned short* __restrict__ Thi,
                                   unsigned short* __restrict__ Tlo,
                                   float* __restrict__ wqv) {
    int q = blockIdx.x * blockDim.x + threadIdx.x;
    if (q >= QPAD) return;
    if (q >= QTOT) {
        for (int c = 0; c < BPADC; ++c) { Bhi[q * BPADC + c] = 0; Blo[q * BPADC + c] = 0; }
        for (int k = 0; k < TROWS; ++k) { Thi[k * QPAD + q] = 0; Tlo[k * QPAD + q] = 0; }
        wqv[q] = 0.f;
        return;
    }
    const double FACT[9] = {1, 1, 2, 6, 24, 120, 720, 5040, 40320};
    int a = q / (NBr * NCr);
    int b = (q / NCr) % NBr;
    int c = q % NCr;
    double alpha = 2.0 * PI_D * a / NAr;
    double gamma = 2.0 * PI_D * c / NCr;

    // Gauss-Legendre node (ascending index b -> descending-guess index jj)
    int jj = NBr - 1 - b;
    double x = cos(PI_D * (jj + 0.75) / (NBr + 0.5));
    double dp = 1.0;
    for (int it = 0; it < 100; ++it) {
        double pa = 1.0, pb = x;
        for (int k = 2; k <= NBr; ++k) {
            double pc = ((2.0 * k - 1.0) * x * pb - (k - 1.0) * pa) / k;
            pa = pb; pb = pc;
        }
        dp = NBr * (x * pb - pa) / (x * x - 1.0);
        x -= pb / dp;
    }
    { // recompute dp at converged node
        double pa = 1.0, pb = x;
        for (int k = 2; k <= NBr; ++k) {
            double pc = ((2.0 * k - 1.0) * x * pb - (k - 1.0) * pa) / k;
            pa = pb; pb = pc;
        }
        dp = NBr * (x * pb - pa) / (x * x - 1.0);
    }
    double wg = 2.0 / ((1.0 - x * x) * dp * dp);
    wqv[q] = (float)(wg / (2.0 * NAr * NCr));
    double beta = acos(x);
    double cb = cos(0.5 * beta), sb = sin(0.5 * beta);

    int off = 0;
    for (int l = 0; l <= 4; ++l) {
        int n = 2 * l + 1;
        double dm[9][9];
        for (int imp = 0; imp < n; ++imp) {
            int mp = imp - l;
            for (int im = 0; im < n; ++im) {
                int m = im - l;
                double pref = sqrt(FACT[l + mp] * FACT[l - mp] * FACT[l + m] * FACT[l - m]);
                double acc = 0.0;
                int k0 = (m - mp > 0) ? (m - mp) : 0;
                int k1 = (l + m < l - mp) ? (l + m) : (l - mp);
                for (int k = k0; k <= k1; ++k) {
                    double den = FACT[l + m - k] * FACT[k] * FACT[l - mp - k] * FACT[mp - m + k];
                    double t = ipow_d(cb, 2 * l + m - mp - 2 * k) * ipow_d(sb, mp - m + 2 * k) / den;
                    acc += ((mp - m + k) & 1) ? -t : t;
                }
                dm[imp][im] = pref * acc;
            }
        }
        double s = sqrt(2.0 * l + 1.0);
        for (int p = 0; p < n; ++p) {
            int muP[2]; double prC[2], piC[2];
            int nsp = urow(p - l, muP, prC, piC);
            for (int qq = 0; qq < n; ++qq) {
                int muQ[2]; double qrC[2], qiC[2];
                int nsq = urow(qq - l, muQ, qrC, qiC);
                double val = 0.0;
                for (int i1 = 0; i1 < nsp; ++i1)
                    for (int i2 = 0; i2 < nsq; ++i2) {
                        double br = qrC[i2], bi = -qiC[i2];        // conj(U[q,nu])
                        double cr = prC[i1] * br - piC[i1] * bi;
                        double ci = prC[i1] * bi + piC[i1] * br;
                        double th = muP[i1] * alpha + muQ[i2] * gamma; // e^{-i th}
                        val += (cr * cos(th) + ci * sin(th)) * dm[l + muP[i1]][l + muQ[i2]];
                    }
                float fv = (float)(s * val);
                int col = off + p * n + qq;
                unsigned short hv = bf16_rn(fv);
                unsigned short lv = bf16_rn(fv - bf16f(hv));
                Bhi[q * BPADC + col] = hv;  Blo[q * BPADC + col] = lv;
                Thi[col * QPAD + q] = hv;   Tlo[col * QPAD + q] = lv;
            }
        }
        off += n * n;
    }
    for (int col = DTOT; col < BPADC; ++col) { Bhi[q * BPADC + col] = 0; Blo[q * BPADC + col] = 0; }
    for (int k = DTOT; k < TROWS; ++k)       { Thi[k * QPAD + q] = 0;    Tlo[k * QPAD + q] = 0; }
}

// ---------------------------------------------------------------------------
// fp32 -> split bf16 activation conversion (with zero padding to kpad)
// ---------------------------------------------------------------------------
__global__ void split_kernel(const float* __restrict__ src,
                             unsigned short* __restrict__ hi,
                             unsigned short* __restrict__ lo,
                             int M, int din, int kpad) {
    int idx = blockIdx.x * 256 + threadIdx.x;
    if (idx >= M * kpad) return;
    int r = idx / kpad, k = idx % kpad;
    float v = (k < din) ? src[(size_t)r * din + k] : 0.f;
    unsigned short h = bf16_rn(v);
    hi[idx] = h;
    lo[idx] = bf16_rn(v - bf16f(h));
}

__global__ void copy_kernel(const float* __restrict__ s, float* __restrict__ d, int n) {
    int i = blockIdx.x * 256 + threadIdx.x;
    if (i < n) d[i] = s[i];
}

// ---------------------------------------------------------------------------
// Fused so3_act: G = (relu(X @ B^T) * wq) @ B      [split-bf16 WMMA]
//   X: M x (KT1*32) split-bf16, T = B^T: TROWS x QPAD, B: QPAD x BPADC
//   4 waves / block, 16 output rows / block, Q loop in tiles of 64
// ---------------------------------------------------------------------------
template <int KT1, int NT>
__global__ __launch_bounds__(128) void act_kernel(
    const unsigned short* __restrict__ Xhi, const unsigned short* __restrict__ Xlo,
    const unsigned short* __restrict__ Thi, const unsigned short* __restrict__ Tlo,
    const unsigned short* __restrict__ Bhi, const unsigned short* __restrict__ Blo,
    const float* __restrict__ wqv, float* __restrict__ G, int dout) {
    constexpr int KPAD = KT1 * 32;
    __shared__ alignas(16) float Fs[16 * 66];

    int lane = threadIdx.x & 31;
    int wid  = threadIdx.x >> 5;
    int mrow = lane & 15;        // A-matrix row / C-matrix column (lane%16)
    int half = lane >> 4;
    int m0   = blockIdx.x * 16;

    // Preload A fragments of the X tile (reused for all 84 Q tiles)
    v4u aHi0[KT1], aHi1[KT1], aLo0[KT1], aLo1[KT1];
    {
        const unsigned short* xh = Xhi + (size_t)(m0 + mrow) * KPAD;
        const unsigned short* xl = Xlo + (size_t)(m0 + mrow) * KPAD;
#pragma unroll
        for (int kt = 0; kt < KT1; ++kt) {
            int kb = kt * 32 + 8 * half;
            aHi0[kt] = *(const v4u*)(xh + kb);
            aHi1[kt] = *(const v4u*)(xh + kb + 16);
            aLo0[kt] = *(const v4u*)(xl + kb);
            aLo1[kt] = *(const v4u*)(xl + kb + 16);
        }
    }

    v8f acc0 = zero8(), acc1 = zero8(), acc2 = zero8();

    for (int qt = 0; qt < NQT; ++qt) {
        int q0 = qt * QT;
        if (qt + 1 < NQT) { // warm L1/L2 with next basis tile
            __builtin_prefetch(Bhi + (size_t)(q0 + QT + lane) * BPADC, 0, 1);
        }
        // ---- stage 1: F(16x16) = X * T for this wave's q sub-tile ----
        v8f c = zero8();
#pragma unroll
        for (int kt = 0; kt < KT1; ++kt) {
            int krow = kt * 32 + 8 * half;
            const unsigned short* th = Thi + (size_t)krow * QPAD + q0 + wid * 16 + mrow;
            const unsigned short* tl = Tlo + (size_t)krow * QPAD + q0 + wid * 16 + mrow;
            v16bf bh, bl;
#pragma unroll
            for (int e = 0; e < 16; ++e) {
                int kk = (e < 8) ? e : e + 8;
                bh[e] = __builtin_bit_cast(__bf16, th[(size_t)kk * QPAD]);
                bl[e] = __builtin_bit_cast(__bf16, tl[(size_t)kk * QPAD]);
            }
            v16bf ah = mk_frag(aHi0[kt], aHi1[kt]);
            v16bf al = mk_frag(aLo0[kt], aLo1[kt]);
            c = wmma_bf16(ah, bh, c);
            c = wmma_bf16(ah, bl, c);
            c = wmma_bf16(al, bh, c);
        }
        // relu + quadrature weight, park in LDS (f32)
        float wqs = wqv[q0 + wid * 16 + mrow];
#pragma unroll
        for (int j = 0; j < 8; ++j) {
            float v = c[j];
            v = v > 0.f ? v : 0.f;
            Fs[(j + 8 * half) * 66 + wid * 16 + mrow] = v * wqs;
        }
        __syncthreads();
        // ---- stage 2: G += F * B,  K = 64 (2 k-tiles), N tiles split over waves
#pragma unroll
        for (int kt2 = 0; kt2 < 2; ++kt2) {
            v8u uh, ul;
#pragma unroll
            for (int j = 0; j < 4; ++j) {
                v2f p0 = *(const v2f*)&Fs[mrow * 66 + kt2 * 32 + 8 * half + 2 * j];
                v2f p1 = *(const v2f*)&Fs[mrow * 66 + kt2 * 32 + 16 + 8 * half + 2 * j];
                unsigned int hw, lw;
                splitpack(p0[0], p0[1], hw, lw);
                uh[j] = hw; ul[j] = lw;
                splitpack(p1[0], p1[1], hw, lw);
                uh[4 + j] = hw; ul[4 + j] = lw;
            }
            v16bf a2h = __builtin_bit_cast(v16bf, uh);
            v16bf a2l = __builtin_bit_cast(v16bf, ul);
            int qrow = q0 + kt2 * 32 + 8 * half;
#pragma unroll
            for (int s = 0; s < 3; ++s) {
                int nt = wid + 4 * s;           // wave-uniform -> EXEC stays full
                if (nt < NT) {
                    const unsigned short* bhp = Thi + (size_t)(nt * 16 + mrow) * QPAD + qrow;
                    const unsigned short* blp = Tlo + (size_t)(nt * 16 + mrow) * QPAD + qrow;
                    v16bf b2h = mk_frag(*(const v4u*)bhp, *(const v4u*)(bhp + 16));
                    v16bf b2l = mk_frag(*(const v4u*)blp, *(const v4u*)(blp + 16));
                    v8f& A = (s == 0) ? acc0 : ((s == 1) ? acc1 : acc2);
                    A = wmma_bf16(a2h, b2h, A);
                    A = wmma_bf16(a2h, b2l, A);
                    A = wmma_bf16(a2l, b2h, A);
                }
            }
        }
        __syncthreads();
    }
    // ---- store G ----
#pragma unroll
    for (int s = 0; s < 3; ++s) {
        int nt = wid + 4 * s;
        if (nt < NT) {
            v8f A = (s == 0) ? acc0 : ((s == 1) ? acc1 : acc2);
            int nn = nt * 16 + mrow;
            if (nn < dout) {
#pragma unroll
                for (int j = 0; j < 8; ++j)
                    G[(size_t)(m0 + j + 8 * half) * dout + nn] = A[j];
            }
        }
    }
}

// ---------------------------------------------------------------------------
// equiv_linear
// ---------------------------------------------------------------------------
struct LinDesc {
    const float* w[5];
    int U[5], W[5], offin[5], offout[5];
    int nl, Fin, Fout, Din, Dout;
};

__global__ void lin_kernel(const float* __restrict__ in, float* __restrict__ out, LinDesc d) {
    int l = blockIdx.y;
    int n = 2 * l + 1;
    int Wl = d.W[l], Ul = d.U[l];
    int tot = BATCHC * d.Fout * Wl * n;
    int idx = blockIdx.x * 256 + threadIdx.x;
    if (idx >= tot) return;
    int i = idx % n; int t = idx / n;
    int w = t % Wl;  t /= Wl;
    int g = t % d.Fout;
    int b = t / d.Fout;
    const float* wp = d.w[l];
    float acc = 0.f;
    for (int f = 0; f < d.Fin; ++f) {
        const float* inr = in + (size_t)(b * d.Fin + f) * d.Din + d.offin[l];
        const float* wr  = wp + (size_t)f * Ul * d.Fout * Wl;
        for (int u = 0; u < Ul; ++u)
            acc += inr[u * n + i] * wr[(size_t)u * d.Fout * Wl + (size_t)g * Wl + w];
    }
    out[(size_t)(b * d.Fout + g) * d.Dout + d.offout[l] + w * n + i] =
        acc * rsqrtf((float)(d.Fin * Ul));
}

// ---------------------------------------------------------------------------
// irrep_bn: one block per mul-channel (muls = so3_muls -> mul_l = n_l = 2l+1)
// ---------------------------------------------------------------------------
__global__ void bn_kernel(const float* __restrict__ in, float* __restrict__ out,
                          const float* __restrict__ weight, const float* __restrict__ bias,
                          int Bsz, int F, int D) {
    int ch = blockIdx.x;
    int l = 0, base = 0, rem = ch;
    for (;;) {
        int mul = 2 * l + 1;
        if (rem < mul) break;
        rem -= mul; base += mul * mul; ++l;
    }
    int n = 2 * l + 1;
    int off = base + rem * n;
    int cnt = Bsz * F * n;
    __shared__ float red[256];

    float mean = 0.f;
    if (l == 0) {
        float s = 0.f;
        for (int t = threadIdx.x; t < cnt; t += 256) {
            int i = t % n, bf = t / n;
            s += in[(size_t)bf * D + off + i];
        }
        red[threadIdx.x] = s; __syncthreads();
        for (int st = 128; st > 0; st >>= 1) {
            if (threadIdx.x < st) red[threadIdx.x] += red[threadIdx.x + st];
            __syncthreads();
        }
        mean = red[0] / (float)cnt;
        __syncthreads();
    }
    float s2 = 0.f;
    for (int t = threadIdx.x; t < cnt; t += 256) {
        int i = t % n, bf = t / n;
        float v = in[(size_t)bf * D + off + i] - mean;
        s2 += v * v;
    }
    red[threadIdx.x] = s2; __syncthreads();
    for (int st = 128; st > 0; st >>= 1) {
        if (threadIdx.x < st) red[threadIdx.x] += red[threadIdx.x + st];
        __syncthreads();
    }
    float norm = red[0] / (float)cnt;
    float scale = rsqrtf(norm + 1e-5f) * weight[ch];
    float b0 = (l == 0) ? bias[0] : 0.f;
    for (int t = threadIdx.x; t < cnt; t += 256) {
        int i = t % n, bf = t / n;
        float v = in[(size_t)bf * D + off + i] - mean;
        out[(size_t)bf * D + off + i] = v * scale + b0;
    }
}

// ---------------------------------------------------------------------------
// Host orchestration
// ---------------------------------------------------------------------------
static void run_lin(const float* in, float* out, void* const* d_in, int base_idx,
                    int Fin, int Fout, const int* mi, const int* mo, int nl,
                    hipStream_t stream) {
    LinDesc ld{};
    int offi = 0, offo = 0, maxwn = 0;
    for (int l = 0; l < nl; ++l) {
        int n = 2 * l + 1;
        ld.w[l] = (const float*)d_in[base_idx + l];
        ld.U[l] = mi[l]; ld.W[l] = mo[l];
        ld.offin[l] = offi; ld.offout[l] = offo;
        offi += mi[l] * n; offo += mo[l] * n;
        if (mo[l] * n > maxwn) maxwn = mo[l] * n;
    }
    ld.nl = nl; ld.Fin = Fin; ld.Fout = Fout; ld.Din = offi; ld.Dout = offo;
    int maxtot = BATCHC * Fout * maxwn;
    dim3 grid((maxtot + 255) / 256, nl);
    lin_kernel<<<grid, 256, 0, stream>>>(in, out, ld);
}

static void run_split(const float* src, unsigned short* hi, unsigned short* lo,
                      int M, int din, int kpad, hipStream_t stream) {
    int tot = M * kpad;
    split_kernel<<<(tot + 255) / 256, 256, 0, stream>>>(src, hi, lo, M, din, kpad);
}

extern "C" void kernel_launch(void* const* d_in, const int* in_sizes, int n_in,
                              void* d_out, int out_size, void* d_ws, size_t ws_size,
                              hipStream_t stream) {
    (void)in_sizes; (void)n_in; (void)out_size; (void)ws_size;
    const float* x = (const float*)d_in[0];
    char* ws = (char*)d_ws;
    float*          wqv = (float*)(ws + OFF_WQ);
    unsigned short* BHI = (unsigned short*)(ws + OFF_BHI);
    unsigned short* BLO = (unsigned short*)(ws + OFF_BLO);
    unsigned short* THI = (unsigned short*)(ws + OFF_THI);
    unsigned short* TLO = (unsigned short*)(ws + OFF_TLO);
    unsigned short* XHI = (unsigned short*)(ws + OFF_XHI);
    unsigned short* XLO = (unsigned short*)(ws + OFF_XLO);
    float* bufA = (float*)(ws + OFF_A);
    float* bufB = (float*)(ws + OFF_B);

    float* outp = (float*)d_out;
    float* lat  = outp + 800;                       // (32,128,35)
    float* fin  = outp + 800 + 32 * 128 * 35;       // (32,1,25)

    static const int s24m[5] = {1, 1, 1, 1, 1};
    static const int so4m[5] = {1, 3, 5, 7, 9};
    static const int so2m[3] = {1, 3, 5};

    build_basis_kernel<<<(QPAD + 63) / 64, 64, 0, stream>>>(BHI, BLO, THI, TLO, wqv);
    copy_kernel<<<(800 + 255) / 256, 256, 0, stream>>>(x, outp, 800);

    // ---- encoder ----
    run_lin(x, bufA, d_in, 1, 1, 32, s24m, so4m, 5, stream);              // enc_lin0
    run_split(bufA, XHI, XLO, 1024, 165, 192, stream);
    act_kernel<6, 11><<<1024 / 16, 128, 0, stream>>>(XHI, XLO, THI, TLO, BHI, BLO, wqv, bufB, 165);
    bn_kernel<<<25, 256, 0, stream>>>(bufB, bufA, (const float*)d_in[29], (const float*)d_in[30], 32, 32, 165);
    run_lin(bufA, bufB, d_in, 6, 32, 64, so4m, so4m, 5, stream);          // enc_lin1
    bn_kernel<<<25, 256, 0, stream>>>(bufB, bufA, (const float*)d_in[31], (const float*)d_in[32], 32, 64, 165);
    run_split(bufA, XHI, XLO, 2048, 165, 192, stream);
    act_kernel<6, 11><<<2048 / 16, 128, 0, stream>>>(XHI, XLO, THI, TLO, BHI, BLO, wqv, bufB, 165);
    run_lin(bufB, bufA, d_in, 11, 64, 128, so4m, so4m, 5, stream);        // enc_lin2
    bn_kernel<<<25, 256, 0, stream>>>(bufA, bufB, (const float*)d_in[33], (const float*)d_in[34], 32, 128, 165);
    run_split(bufB, XHI, XLO, 4096, 165, 192, stream);
    act_kernel<6, 3><<<4096 / 16, 128, 0, stream>>>(XHI, XLO, THI, TLO, BHI, BLO, wqv, lat, 35);   // -> latent

    // ---- decoder ----
    run_lin(lat, bufA, d_in, 16, 128, 64, so2m, so2m, 3, stream);         // dec_lin0
    bn_kernel<<<9, 256, 0, stream>>>(bufA, bufB, (const float*)d_in[35], (const float*)d_in[36], 32, 64, 35);
    run_split(bufB, XHI, XLO, 2048, 35, 64, stream);
    act_kernel<2, 11><<<2048 / 16, 128, 0, stream>>>(XHI, XLO, THI, TLO, BHI, BLO, wqv, bufA, 165);
    run_lin(bufA, bufB, d_in, 19, 64, 32, so4m, so4m, 5, stream);         // dec_lin1
    bn_kernel<<<25, 256, 0, stream>>>(bufB, bufA, (const float*)d_in[37], (const float*)d_in[38], 32, 32, 165);
    run_split(bufA, XHI, XLO, 1024, 165, 192, stream);
    act_kernel<6, 11><<<1024 / 16, 128, 0, stream>>>(XHI, XLO, THI, TLO, BHI, BLO, wqv, bufB, 165);
    run_lin(bufB, fin, d_in, 24, 32, 1, so4m, s24m, 5, stream);           // dec_lin2
}
